// Prototype_Sampler_Network_7310034337851
// MI455X (gfx1250) — matrix operand order repaired and checked by reading, not compiled, not run on hardware
//
#include <hip/hip_runtime.h>
#include <hip/hip_bf16.h>

typedef __attribute__((ext_vector_type(16))) _Float16 v16h;
typedef __attribute__((ext_vector_type(8)))  _Float16 v8h;
typedef __attribute__((ext_vector_type(8)))  float    v8f;
typedef __attribute__((ext_vector_type(4)))  float    v4f;

// ---------------------------------------------------------------- WMMA helpers

__device__ static inline v8f wmma_f16(v16h a, v16h b, v8f c) {
  return __builtin_amdgcn_wmma_f32_16x16x32_f16(false, a, false, b, (short)0, c,
                                                false, false);
}

// A tile (16 rows x 32 cols) from a row-major f16 matrix, src points at
// (row0, k0), stride ld (halfs).  ISA layout: lane = M (dup over halves);
// element i = K = i + 8h (i<8), K = 16 + (i-8) + 8h (i>=8)  -> two contiguous
// 16-byte runs per lane => 2x global_load_b128.
__device__ static inline v16h load_A16(const _Float16* __restrict__ src, int ld,
                                       int lane) {
  int m = lane & 15, h = lane >> 4;
  const _Float16* rp = src + (size_t)m * ld + 8 * h;
  v8h lo = *(const v8h*)rp;
  v8h hi = *(const v8h*)(rp + 16);
  return __builtin_shufflevector(lo, hi, 0, 1, 2, 3, 4, 5, 6, 7, 8, 9, 10, 11, 12,
                                 13, 14, 15);
}

// B tile (32 x 16), B[k][n] = srcT[n][k]  (srcT row-major, stride ld).
// ISA layout: lane = N, element i = K = i + 8h (i<8), K = 16 + (i-8) + 8h (i>=8)
// -> two contiguous 16-byte runs per lane => 2x global_load_b128.
__device__ static inline v16h load_B_T(const _Float16* __restrict__ srcT, int ld,
                                       int lane) {
  int n = lane & 15, h = lane >> 4;
  const _Float16* rp = srcT + (size_t)n * ld + 8 * h;
  v8h lo = *(const v8h*)rp;
  v8h hi = *(const v8h*)(rp + 16);
  return __builtin_shufflevector(lo, hi, 0, 1, 2, 3, 4, 5, 6, 7, 8, 9, 10, 11, 12,
                                 13, 14, 15);
}

// ---------------------------------------------------------------- weight prep

// Transpose + convert the small f32 weights to f16 so every WMMA B tile is a
// contiguous b128 load.  tw/pw/gw: (64x32) -> T (32x64);  ww: (32x64) -> T (64x32)
__global__ void k_prep(const float* __restrict__ tw, const float* __restrict__ pw,
                       const float* __restrict__ gw, const float* __restrict__ ww,
                       _Float16* __restrict__ twT, _Float16* __restrict__ pwT,
                       _Float16* __restrict__ gwT, _Float16* __restrict__ wwT) {
  for (int i = threadIdx.x; i < 2048; i += 256) {
    int k = i >> 5, n = i & 31;  // tw[i] = tw[k][n]
    twT[n * 64 + k] = (_Float16)tw[i];
    pwT[n * 64 + k] = (_Float16)pw[i];
    gwT[n * 64 + k] = (_Float16)gw[i];
    int k2 = i >> 6, n2 = i & 63;  // ww[i] = ww[k2][n2]
    wwT[n2 * 32 + k2] = (_Float16)ww[i];
  }
}

// ---------------------------------------------------------------- BN1 stats

__global__ void k_bn1_partial(const float* __restrict__ x, float* __restrict__ p1,
                              float* __restrict__ p2) {
  __shared__ float r1[256], r2[256];
  int c = threadIdx.x & 63;
  int rg = threadIdx.x >> 6;  // 0..3
  int row0 = blockIdx.x * 128;
  float s1 = 0.f, s2 = 0.f;
  for (int r = rg; r < 128; r += 4) {
    float v = x[(size_t)(row0 + r) * 64 + c];
    s1 += v;
    s2 += v * v;
  }
  r1[threadIdx.x] = s1;
  r2[threadIdx.x] = s2;
  __syncthreads();
  if (rg == 0) {
    s1 = r1[c] + r1[c + 64] + r1[c + 128] + r1[c + 192];
    s2 = r2[c] + r2[c + 64] + r2[c + 128] + r2[c + 192];
    p1[blockIdx.x * 64 + c] = s1;
    p2[blockIdx.x * 64 + c] = s2;
  }
}

// shared by BN1 and BN3: scale = g*rsqrt(var+eps), shift = b - mean*scale
__global__ void k_bn_final(const float* __restrict__ p1, const float* __restrict__ p2,
                           const float* __restrict__ g, const float* __restrict__ b,
                           float* __restrict__ sc, float* __restrict__ sh, int npart,
                           float count) {
  int c = threadIdx.x;  // 64 threads
  float s1 = 0.f, s2 = 0.f;
  for (int i = 0; i < npart; ++i) {
    s1 += p1[i * 64 + c];
    s2 += p2[i * 64 + c];
  }
  float m = s1 / count;
  float v = s2 / count - m * m;
  float scale = g[c] * rsqrtf(v + 1e-5f);
  sc[c] = scale;
  sh[c] = b[c] - m * scale;
}

__global__ void k_bn1_apply(const float* __restrict__ x, const float* __restrict__ sc,
                            const float* __restrict__ sh, _Float16* __restrict__ f16f) {
  size_t i = (size_t)blockIdx.x * blockDim.x + threadIdx.x;  // 1048576 total
  int c = (int)(i & 63);
  float v = x[i] * sc[c] + sh[c];
  f16f[i] = (_Float16)(v > 0.f ? v : 0.f);
}

// ---------------------------------------------------------------- projections

// t = f@theta+tb ; p = f@phi+pb (row-major 16384x32 f16);
// gg = f@gw+gb stored TRANSPOSED: ggT[(b*32+d)*4096 + m].
__global__ void k_proj(const _Float16* __restrict__ f, const _Float16* __restrict__ twT,
                       const float* __restrict__ tb, const _Float16* __restrict__ pwT,
                       const float* __restrict__ pb, const _Float16* __restrict__ gwT,
                       const float* __restrict__ gb, _Float16* __restrict__ t16,
                       _Float16* __restrict__ p16, _Float16* __restrict__ ggT) {
  int lane = threadIdx.x & 31;
  int tile = blockIdx.x * 8 + (threadIdx.x >> 5);  // 0..1023  (16 rows each)
  int q = lane & 15, h = lane >> 4;
  const _Float16* arow = f + (size_t)tile * 16 * 64;
  v16h a0 = load_A16(arow, 64, lane);
  v16h a1 = load_A16(arow + 32, 64, lane);
#pragma unroll
  for (int t = 0; t < 2; ++t) {
    // theta -> t16 (row-major)
    {
      v8f acc = {};
      acc = wmma_f16(a0, load_B_T(twT + (size_t)(16 * t) * 64, 64, lane), acc);
      acc = wmma_f16(a1, load_B_T(twT + (size_t)(16 * t) * 64 + 32, 64, lane), acc);
      float bias = tb[16 * t + q];
#pragma unroll
      for (int r = 0; r < 8; ++r) {
        size_t row = (size_t)tile * 16 + r + 8 * h;
        t16[row * 32 + 16 * t + q] = (_Float16)(acc[r] + bias);
      }
    }
    // phi -> p16 (row-major)
    {
      v8f acc = {};
      acc = wmma_f16(a0, load_B_T(pwT + (size_t)(16 * t) * 64, 64, lane), acc);
      acc = wmma_f16(a1, load_B_T(pwT + (size_t)(16 * t) * 64 + 32, 64, lane), acc);
      float bias = pb[16 * t + q];
#pragma unroll
      for (int r = 0; r < 8; ++r) {
        size_t row = (size_t)tile * 16 + r + 8 * h;
        p16[row * 32 + 16 * t + q] = (_Float16)(acc[r] + bias);
      }
    }
    // g -> ggT (transposed, vector store)
    {
      v8f acc = {};
      acc = wmma_f16(a0, load_B_T(gwT + (size_t)(16 * t) * 64, 64, lane), acc);
      acc = wmma_f16(a1, load_B_T(gwT + (size_t)(16 * t) * 64 + 32, 64, lane), acc);
      float bias = gb[16 * t + q];
      v8h ov;
#pragma unroll
      for (int r = 0; r < 8; ++r) ov[r] = (_Float16)(acc[r] + bias);
      int bb = tile >> 8;
      int n0 = (tile & 255) * 16 + 8 * h;  // 16B aligned
      *(v8h*)(ggT + ((size_t)bb * 32 + 16 * t + q) * 4096 + n0) = ov;
    }
  }
}

// ---------------------------------------------------------------- flash attention
// Split-K: 4 waves per 16-query tile, each covering 1024 keys; block = 8 waves
// = 2 tiles.  Partial (max,sum,acc) merged via LDS (log-sum-exp recombine).

__global__ void k_flash(const _Float16* __restrict__ t16,
                        const _Float16* __restrict__ p16,
                        const _Float16* __restrict__ ggT, _Float16* __restrict__ o16) {
  __shared__ float Lm[2][4][16], Ls[2][4][16];
  __shared__ float Lacc[2][4][32][16];
  int lane = threadIdx.x & 31;
  int warp = threadIdx.x >> 5;  // 0..7
  int tloc = warp >> 2;         // tile in block: 0..1
  int sp = warp & 3;            // key split: 0..3
  int tile = blockIdx.x * 2 + tloc;  // 0..1023
  int b = tile >> 8;
  int q0 = (tile & 255) * 16;
  int q = lane & 15, h = lane >> 4;
  size_t base = (size_t)b * 4096;

  v16h Bq = load_B_T(t16 + (base + q0) * 32, 32, lane);  // loop-invariant

  float run_max = -3.0e38f, run_sum = 0.f;
  v8f acc0 = {}, acc1 = {};

  for (int m0 = sp * 1024; m0 < (sp + 1) * 1024; m0 += 32) {
    v16h A0 = load_A16(p16 + (base + m0) * 32, 32, lane);
    v16h A1 = load_A16(p16 + (base + m0 + 16) * 32, 32, lane);
    v8f s0 = {}, s1 = {};
    s0 = wmma_f16(A0, Bq, s0);  // keys m0..m0+15  x queries
    s1 = wmma_f16(A1, Bq, s1);  // keys m0+16..m0+31

    float tmax = s0[0];
#pragma unroll
    for (int r = 1; r < 8; ++r) tmax = fmaxf(tmax, s0[r]);
#pragma unroll
    for (int r = 0; r < 8; ++r) tmax = fmaxf(tmax, s1[r]);
    tmax = fmaxf(tmax, __shfl_xor(tmax, 16));  // merge key-halves (same query)

    float nmax = fmaxf(run_max, tmax);
    float factor = __expf(run_max - nmax);
    float e0[8], e1[8], lsum = 0.f;
#pragma unroll
    for (int r = 0; r < 8; ++r) {
      e0[r] = __expf(s0[r] - nmax);
      e1[r] = __expf(s1[r] - nmax);
      lsum += e0[r] + e1[r];
    }
    lsum += __shfl_xor(lsum, 16);
    run_sum = run_sum * factor + lsum;
    run_max = nmax;

    // P tile in A-layout: this lane's e-values are exactly its A K-slots.
    v16h Ap;
#pragma unroll
    for (int r = 0; r < 8; ++r) {
      Ap[r] = (_Float16)e0[r];
      Ap[8 + r] = (_Float16)e1[r];
    }

    // rescale accumulators: element r is query (r + 8h) -> factor at lane r+8h
#pragma unroll
    for (int r = 0; r < 8; ++r) {
      float fr = __shfl(factor, r + 8 * h);
      acc0[r] *= fr;
      acc1[r] *= fr;
    }
    v16h Bg0 = load_B_T(ggT + ((size_t)b * 32 + 0) * 4096 + m0, 4096, lane);
    v16h Bg1 = load_B_T(ggT + ((size_t)b * 32 + 16) * 4096 + m0, 4096, lane);
    acc0 = wmma_f16(Ap, Bg0, acc0);
    acc1 = wmma_f16(Ap, Bg1, acc1);
  }

  // publish partials
  if (h == 0) {
    Lm[tloc][sp][q] = run_max;
    Ls[tloc][sp][q] = run_sum;
  }
#pragma unroll
  for (int r = 0; r < 8; ++r) {
    Lacc[tloc][sp][lane][r] = acc0[r];
    Lacc[tloc][sp][lane][r + 8] = acc1[r];
  }
  __syncthreads();

  if (sp == 0) {
    // combined max / sum per query (lane-wise; both halves identical)
    float M = run_max;
#pragma unroll
    for (int w = 1; w < 4; ++w) M = fmaxf(M, Lm[tloc][w][q]);
    float S = run_sum * __expf(run_max - M);
#pragma unroll
    for (int w = 1; w < 4; ++w) S += Ls[tloc][w][q] * __expf(Lm[tloc][w][q] - M);

    float f0 = __expf(run_max - M);
#pragma unroll
    for (int r = 0; r < 8; ++r) {
      float fr = __shfl(f0, r + 8 * h);
      acc0[r] *= fr;
      acc1[r] *= fr;
    }
#pragma unroll
    for (int w = 1; w < 4; ++w) {
      float fw = __expf(Lm[tloc][w][q] - M);
#pragma unroll
      for (int r = 0; r < 8; ++r) {
        float fr = __shfl(fw, r + 8 * h);
        acc0[r] += Lacc[tloc][w][lane][r] * fr;
        acc1[r] += Lacc[tloc][w][lane][r + 8] * fr;
      }
    }
#pragma unroll
    for (int r = 0; r < 8; ++r) {
      float sr = __shfl(S, r + 8 * h);
      float inv = 1.f / sr;
      size_t row = base + q0 + r + 8 * h;
      o16[row * 32 + q] = (_Float16)(acc0[r] * inv);
      o16[row * 32 + 16 + q] = (_Float16)(acc1[r] * inv);
    }
  }
}

// ---------------------------------------------------------------- att_fea = x + o@w_w + w_b

__global__ void k_attfea(const float* __restrict__ x, const _Float16* __restrict__ o16,
                         const _Float16* __restrict__ wwT, const float* __restrict__ wb,
                         float* __restrict__ af, _Float16* __restrict__ af16) {
  int lane = threadIdx.x & 31;
  int tile = blockIdx.x * 8 + (threadIdx.x >> 5);
  int q = lane & 15, h = lane >> 4;
  v16h A = load_A16(o16 + (size_t)tile * 16 * 32, 32, lane);
#pragma unroll
  for (int t = 0; t < 4; ++t) {
    v16h B = load_B_T(wwT + (size_t)(16 * t) * 32, 32, lane);
    v8f acc = {};
    acc = wmma_f16(A, B, acc);
    int col = 16 * t + q;
    float bias = wb[col];
#pragma unroll
    for (int r = 0; r < 8; ++r) {
      size_t row = (size_t)tile * 16 + r + 8 * h;
      float v = x[row * 64 + col] + acc[r] + bias;
      af[row * 64 + col] = v;
      af16[row * 64 + col] = (_Float16)v;
    }
  }
}

__global__ void k_sq(const float* __restrict__ af, float* __restrict__ sq) {
  size_t row = (size_t)blockIdx.x * blockDim.x + threadIdx.x;
  const v4f* rp = (const v4f*)(af + row * 64);
  float s = 0.f;
#pragma unroll
  for (int i = 0; i < 16; ++i) {
    v4f v = rp[i];
    s += v.x * v.x + v.y * v.y + v.z * v.z + v.w * v.w;
  }
  sq[row] = s;
}

// ---------------------------------------------------------------- fused pdist + top-10
// Split-K like k_flash: 4 waves per query tile, merge top-10 lists via LDS.

__device__ static inline void topk_insert(float (&tv)[10], int (&ti)[10], float d,
                                          int key) {
  if (d <= tv[0]) return;
  int pos = 0;
#pragma unroll
  for (int j = 1; j < 10; ++j)
    if (d > tv[j]) pos = j;
#pragma unroll
  for (int j = 0; j < 9; ++j)
    if (j < pos) {
      tv[j] = tv[j + 1];
      ti[j] = ti[j + 1];
    }
#pragma unroll
  for (int j = 0; j < 10; ++j)
    if (j == pos) {
      tv[j] = d;
      ti[j] = key;
    }
}

__global__ void k_topk(const _Float16* __restrict__ af16, const float* __restrict__ sq,
                       int* __restrict__ idxout) {
  __shared__ float LV[2][4][16][10];
  __shared__ int LI[2][4][16][10];
  int lane = threadIdx.x & 31;
  int warp = threadIdx.x >> 5;
  int tloc = warp >> 2;
  int sp = warp & 3;
  int tile = blockIdx.x * 2 + tloc;
  int b = tile >> 8;
  int q0 = (tile & 255) * 16;
  int q = lane & 15, h = lane >> 4;
  size_t base = (size_t)b * 4096;

  v16h Bq0 = load_B_T(af16 + (base + q0) * 64, 64, lane);
  v16h Bq1 = load_B_T(af16 + (base + q0) * 64 + 32, 64, lane);
  float sqq = sq[base + q0 + q];

  float tv[10];
  int ti[10];
#pragma unroll
  for (int j = 0; j < 10; ++j) {
    tv[j] = -3.0e38f;
    ti[j] = 0;
  }

  for (int m0 = sp * 1024; m0 < (sp + 1) * 1024; m0 += 16) {
    v16h A0 = load_A16(af16 + (base + m0) * 64, 64, lane);
    v16h A1 = load_A16(af16 + (base + m0) * 64 + 32, 64, lane);
    v8f acc = {};
    acc = wmma_f16(A0, Bq0, acc);
    acc = wmma_f16(A1, Bq1, acc);
#pragma unroll
    for (int r = 0; r < 8; ++r) {
      int key = m0 + r + 8 * h;
      float d = 2.f * acc[r] - sqq - sq[base + key];
      topk_insert(tv, ti, d, key);
    }
  }

  // merge the two key-halves within the wave (snapshot first, then insert)
  float ov[10];
  int oi[10];
#pragma unroll
  for (int j = 0; j < 10; ++j) {
    ov[j] = __shfl_xor(tv[j], 16);
    oi[j] = __shfl_xor(ti[j], 16);
  }
#pragma unroll
  for (int j = 0; j < 10; ++j) topk_insert(tv, ti, ov[j], oi[j]);

  // publish per-split lists, merge across the 4 splits
  if (h == 0) {
#pragma unroll
    for (int j = 0; j < 10; ++j) {
      LV[tloc][sp][q][j] = tv[j];
      LI[tloc][sp][q][j] = ti[j];
    }
  }
  __syncthreads();
  if (sp == 0 && h == 0) {
#pragma unroll
    for (int w = 1; w < 4; ++w)
#pragma unroll
      for (int j = 0; j < 10; ++j)
        topk_insert(tv, ti, LV[tloc][w][q][j], LI[tloc][w][q][j]);
#pragma unroll
    for (int j = 0; j < 10; ++j)
      idxout[(base + q0 + q) * 10 + j] = ti[9 - j];  // descending
  }
}

// ---------------------------------------------------------------- conv3 stats (pass 1)

__global__ void k_conv_stats(const float* __restrict__ af, const int* __restrict__ idx,
                             const float* __restrict__ W, float* __restrict__ p1,
                             float* __restrict__ p2) {
  __shared__ float rows[11][64];
  int c = threadIdx.x;  // 64 threads
  float s1 = 0.f, s2 = 0.f;
  for (int pt = 0; pt < 8; ++pt) {
    int point = blockIdx.x * 8 + pt;  // 0..16383
    int bb = point >> 12;
    __syncthreads();
    rows[0][c] = af[(size_t)point * 64 + c];
#pragma unroll
    for (int k = 0; k < 10; ++k) {
      int nb = idx[(size_t)point * 10 + k];
      rows[1 + k][c] = af[((size_t)bb * 4096 + nb) * 64 + c];
    }
    __syncthreads();
    float hk[10];
#pragma unroll
    for (int k = 0; k < 10; ++k) hk[k] = 0.f;
    float common = 0.f;
    for (int c2 = 0; c2 < 64; ++c2) {
      float w1 = W[c * 128 + c2];
      float w2 = W[c * 128 + 64 + c2];
      float a = rows[0][c2];
      common += a * (w2 - w1);
#pragma unroll
      for (int k = 0; k < 10; ++k) hk[k] += rows[1 + k][c2] * w1;
    }
#pragma unroll
    for (int k = 0; k < 10; ++k) {
      float hv = hk[k] + common;
      s1 += hv;
      s2 += hv * hv;
    }
  }
  p1[blockIdx.x * 64 + c] = s1;
  p2[blockIdx.x * 64 + c] = s2;
}

// ---------------------------------------------------------------- conv3 apply + max + fc (pass 2)

__global__ void k_final(const float* __restrict__ af, const int* __restrict__ idx,
                        const float* __restrict__ W, const float* __restrict__ sc3,
                        const float* __restrict__ sh3, const float* __restrict__ fcw,
                        const float* __restrict__ fcb, float* __restrict__ out_s,
                        float* __restrict__ out_f) {
  __shared__ float rows[11][64];
  __shared__ float red[64];
  __shared__ float sval;
  int c = threadIdx.x;
  int point = blockIdx.x;  // 0..16383
  int bb = point >> 12;
  rows[0][c] = af[(size_t)point * 64 + c];
#pragma unroll
  for (int k = 0; k < 10; ++k) {
    int nb = idx[(size_t)point * 10 + k];
    rows[1 + k][c] = af[((size_t)bb * 4096 + nb) * 64 + c];
  }
  __syncthreads();
  float hk[10];
#pragma unroll
  for (int k = 0; k < 10; ++k) hk[k] = 0.f;
  float common = 0.f;
  for (int c2 = 0; c2 < 64; ++c2) {
    float w1 = W[c * 128 + c2];
    float w2 = W[c * 128 + 64 + c2];
    float a = rows[0][c2];
    common += a * (w2 - w1);
#pragma unroll
    for (int k = 0; k < 10; ++k) hk[k] += rows[1 + k][c2] * w1;
  }
  float scale = sc3[c], shift = sh3[c];
  float gp = -3.0e38f;
#pragma unroll
  for (int k = 0; k < 10; ++k) {
    float v = (hk[k] + common) * scale + shift;
    v = v > 0.f ? v : 0.2f * v;  // leaky relu 0.2
    gp = fmaxf(gp, v);
  }
  red[c] = gp * fcw[c];
  __syncthreads();
  for (int s = 32; s > 0; s >>= 1) {
    if (c < s) red[c] += red[c + s];
    __syncthreads();
  }
  if (c == 0) {
    float z = red[0] + fcb[0];
    float sv = 1.f / (1.f + __expf(-z));
    sval = sv;
    out_s[point] = sv;
  }
  __syncthreads();
  out_f[(size_t)point * 64 + c] = sval * rows[0][c];
}

// ---------------------------------------------------------------- host launcher

extern "C" void kernel_launch(void* const* d_in, const int* in_sizes, int n_in,
                              void* d_out, int out_size, void* d_ws, size_t ws_size,
                              hipStream_t stream) {
  (void)in_sizes;
  (void)n_in;
  (void)out_size;
  (void)ws_size;
  const float* x = (const float*)d_in[0];
  const float* bn1g = (const float*)d_in[1];
  const float* bn1b = (const float*)d_in[2];
  const float* tw = (const float*)d_in[3];
  const float* tb = (const float*)d_in[4];
  const float* pw = (const float*)d_in[5];
  const float* pb = (const float*)d_in[6];
  const float* gw = (const float*)d_in[7];
  const float* gb = (const float*)d_in[8];
  const float* ww = (const float*)d_in[9];
  const float* wb = (const float*)d_in[10];
  const float* c3w = (const float*)d_in[11];
  const float* bn3g = (const float*)d_in[12];
  const float* bn3b = (const float*)d_in[13];
  const float* fcw = (const float*)d_in[14];
  const float* fcb = (const float*)d_in[15];

  char* w = (char*)d_ws;
  size_t off = 0;
  auto alloc = [&](size_t bytes) -> void* {
    void* p = w + off;
    off += (bytes + 255) & ~(size_t)255;
    return p;
  };
  float* bn1p1 = (float*)alloc(128 * 64 * 4);
  float* bn1p2 = (float*)alloc(128 * 64 * 4);
  float* bn1sc = (float*)alloc(64 * 4);
  float* bn1sh = (float*)alloc(64 * 4);
  _Float16* twT = (_Float16*)alloc(2048 * 2);
  _Float16* pwT = (_Float16*)alloc(2048 * 2);
  _Float16* gwT = (_Float16*)alloc(2048 * 2);
  _Float16* wwT = (_Float16*)alloc(2048 * 2);
  _Float16* f16f = (_Float16*)alloc((size_t)16384 * 64 * 2);
  _Float16* t16 = (_Float16*)alloc((size_t)16384 * 32 * 2);
  _Float16* p16 = (_Float16*)alloc((size_t)16384 * 32 * 2);
  _Float16* ggT = (_Float16*)alloc((size_t)16384 * 32 * 2);
  _Float16* o16 = (_Float16*)alloc((size_t)16384 * 32 * 2);
  float* af = (float*)alloc((size_t)16384 * 64 * 4);
  _Float16* af16 = (_Float16*)alloc((size_t)16384 * 64 * 2);
  float* sqb = (float*)alloc(16384 * 4);
  int* knn = (int*)alloc((size_t)16384 * 10 * 4);
  float* c3p1 = (float*)alloc((size_t)2048 * 64 * 4);
  float* c3p2 = (float*)alloc((size_t)2048 * 64 * 4);
  float* bn3sc = (float*)alloc(64 * 4);
  float* bn3sh = (float*)alloc(64 * 4);

  float* out_s = (float*)d_out;
  float* out_f = out_s + 16384;

  k_prep<<<1, 256, 0, stream>>>(tw, pw, gw, ww, twT, pwT, gwT, wwT);
  k_bn1_partial<<<128, 256, 0, stream>>>(x, bn1p1, bn1p2);
  k_bn_final<<<1, 64, 0, stream>>>(bn1p1, bn1p2, bn1g, bn1b, bn1sc, bn1sh, 128,
                                   16384.f);
  k_bn1_apply<<<4096, 256, 0, stream>>>(x, bn1sc, bn1sh, f16f);
  k_proj<<<128, 256, 0, stream>>>(f16f, twT, tb, pwT, pb, gwT, gb, t16, p16, ggT);
  k_flash<<<512, 256, 0, stream>>>(t16, p16, ggT, o16);
  k_attfea<<<128, 256, 0, stream>>>(x, o16, wwT, wb, af, af16);
  k_sq<<<64, 256, 0, stream>>>(af, sqb);
  k_topk<<<512, 256, 0, stream>>>(af16, sqb, knn);
  k_conv_stats<<<2048, 64, 0, stream>>>(af, knn, c3w, c3p1, c3p2);
  k_bn_final<<<1, 64, 0, stream>>>(c3p1, c3p2, bn3g, bn3b, bn3sc, bn3sh, 2048,
                                   163840.f);
  k_final<<<16384, 64, 0, stream>>>(af, knn, c3w, bn3sc, bn3sh, fcw, fcb, out_s,
                                    out_f);
}
